// MyOwnDSSNet_53824530153651
// MI455X (gfx1250) — matrix operand order repaired and checked by reading, not compiled, not run on hardware
//
#include <hip/hip_runtime.h>
#include <math.h>

// Problem constants (match reference)
#define NN      100000
#define EE      3200000
#define DDIM    32
#define KSTEPS  4
#define ALPHA_C 0.5f
#define GAMMA_C 0.9f
#define KPAD    96          // edge layer-1 K (66) padded to 3 chunks of 32
#define KGRU    160         // GRU K (131) padded to 5 chunks of 32

typedef __bf16 bf16_t;
typedef bf16_t       v2bf  __attribute__((ext_vector_type(2)));
typedef bf16_t       v16bf __attribute__((ext_vector_type(16)));
typedef float        v8f   __attribute__((ext_vector_type(8)));
typedef unsigned int v8u   __attribute__((ext_vector_type(8)));

// per-wave LDS ordering is in-order on CDNA5; fence only the compiler
#define FENCE() do { __builtin_amdgcn_wave_barrier(); asm volatile("" ::: "memory"); } while (0)

// ---------------- bf16 helpers (native converts -> v_cvt_*bf16*) ----------------
__device__ __forceinline__ unsigned short f2bf(float f) {
  bf16_t h = (bf16_t)f;
  return __builtin_bit_cast(unsigned short, h);
}
__device__ __forceinline__ unsigned int pack2(float a, float b) {
  v2bf t; t[0] = (bf16_t)a; t[1] = (bf16_t)b;
  return __builtin_bit_cast(unsigned int, t);
}

// ---------------- WMMA wrapper ----------------
__device__ __forceinline__ v8f wmma_bf16(v16bf a, v16bf b, v8f c) {
  return __builtin_amdgcn_wmma_f32_16x16x32_bf16(
      false, a, false, b, (short)0, c, false, false);
}

// A fragment (16xK bf16, row-major in LDS, row stride S elements).
// ISA 7.12.2: lane m=L%16, half=L/16; VGPR v<4: K=half*8+2v; v>=4: K=16+half*8+2(v-4).
__device__ __forceinline__ v16bf load_afrag(const unsigned int* lds, int S, int kc, int lane) {
  int m = lane & 15, half = lane >> 4;
  v8u t;
#pragma unroll
  for (int v = 0; v < 8; ++v) {
    int kb = (v < 4) ? (half * 8 + v * 2) : (16 + half * 8 + (v - 4) * 2);
    t[v] = lds[(m * S + kc * 32 + kb) >> 1];
  }
  return __builtin_bit_cast(v16bf, t);
}

// B fragment from global bf16 weights stored row-major [Nout][KS].
// Layout: lane n=L%16 (+16*nt), half=L/16; VGPR v<4: K = kc*32 + half*8 + 2v; v>=4: K = kc*32 + 16 + half*8 + 2(v-4).
__device__ __forceinline__ v16bf load_bfrag(const unsigned short* w, int KS, int kc, int nt, int lane) {
  int n = (lane & 15) + nt * 16;
  int half = lane >> 4;
  const unsigned int* w32 = (const unsigned int*)w;
  v8u t;
#pragma unroll
  for (int v = 0; v < 8; ++v) {
    int k = kc * 32 + ((v < 4) ? (half * 8 + v * 2) : (16 + half * 8 + (v - 4) * 2));
    t[v] = w32[(n * KS + k) >> 1];
  }
  return __builtin_bit_cast(v16bf, t);
}

// ---------------- prep: bf16 weight images ----------------
__global__ void k_prep_weights(const float* __restrict__ toW1, const float* __restrict__ frW1,
                               const float* __restrict__ toW2, const float* __restrict__ frW2,
                               const float* __restrict__ lpW1, const float* __restrict__ lpW2,
                               const float* __restrict__ Wih,  const float* __restrict__ dW1,
                               unsigned short* __restrict__ w1to, unsigned short* __restrict__ w1fr,
                               unsigned short* __restrict__ w2to, unsigned short* __restrict__ w2fr,
                               unsigned short* __restrict__ wlp1, unsigned short* __restrict__ wlp2,
                               unsigned short* __restrict__ wgru, unsigned short* __restrict__ wdec) {
  int t = blockIdx.x * blockDim.x + threadIdx.x;
  if (t < 32 * KPAD) {
    int n = t / KPAD, k = t % KPAD;
    w1to[n * KPAD + k] = f2bf((k < 66) ? toW1[n * 66 + k] : 0.0f);
    // shared edge A tile is [h_dst | h_src | ea]; Phi_from wants [h_src | h_dst | ea]
    int ks = (k < 32) ? (k + 32) : ((k < 64) ? (k - 32) : k);
    w1fr[n * KPAD + k] = f2bf((k < 66) ? frW1[n * 66 + ks] : 0.0f);
  }
  if (t < 32 * 32) {
    w2to[t] = f2bf(toW2[t]);
    w2fr[t] = f2bf(frW2[t]);
    wlp2[t] = f2bf(lpW2[t]);
    wdec[t] = f2bf(dW1[t]);
  }
  if (t < 32 * 64) {  // lp layer1 folded: [H,H,lc] @ W.T == [H,lc] @ Weff.T
    int n = t / 64, k = t % 64;
    float v = (k < 32) ? (lpW1[n * 65 + k] + lpW1[n * 65 + 32 + k])
                       : ((k == 32) ? lpW1[n * 65 + 64] : 0.0f);
    wlp1[n * 64 + k] = f2bf(v);
  }
  if (t < 96 * KGRU) {
    int n = t / KGRU, k = t % KGRU;
    wgru[n * KGRU + k] = f2bf((k < 131) ? Wih[n * 131 + k] : 0.0f);
  }
}

// ---------------- init / degree / zero ----------------
__global__ void k_init(float* __restrict__ H, float* __restrict__ c_dst,
                       float* __restrict__ c_src, float* __restrict__ diag,
                       float* __restrict__ lossbuf) {
  int t = blockIdx.x * blockDim.x + threadIdx.x;
  if (t < NN * DDIM) H[t] = 0.0f;
  if (t < NN) { c_dst[t] = 0.0f; c_src[t] = 0.0f; diag[t] = 1.0f; }
  if (t < 8) lossbuf[t] = 0.0f;
}

__global__ void k_degree(const int* __restrict__ eidx, const float* __restrict__ ea,
                         float* __restrict__ c_dst, float* __restrict__ c_src,
                         float* __restrict__ diag) {
  int e = blockIdx.x * blockDim.x + threadIdx.x;
  if (e >= EE) return;
  int si = eidx[e], di = eidx[EE + e];
  if (si != di) {
    atomicAdd(c_dst + di, 1.0f);
    atomicAdd(c_src + si, 1.0f);
  } else {
    atomicAdd(diag + si, ea[e * 2]);
  }
}

__global__ void k_zero_acc(float* __restrict__ accto, float* __restrict__ accfr) {
  int t = blockIdx.x * blockDim.x + threadIdx.x;
  if (t < NN * DDIM) { accto[t] = 0.0f; accfr[t] = 0.0f; }
}

// ---------------- edge MLP core: 2x WMMA layers + atomic scatter ----------------
__device__ __forceinline__ void mlp_scatter(v16bf a0, v16bf a1, v16bf a2,
                                            const v16bf (&B1)[3][2], const v16bf (&B2)[2],
                                            const float (&bb1)[2], const float (&bb2)[2],
                                            unsigned int* stZ,
                                            const int* sAgg, const int* sD, const int* sS,
                                            float* __restrict__ acc, int lane) {
  int n0 = lane & 15, half = lane >> 4;
#pragma unroll
  for (int nt = 0; nt < 2; ++nt) {
    v8f cc = {};
    cc = wmma_bf16(a0, B1[0][nt], cc);
    cc = wmma_bf16(a1, B1[1][nt], cc);
    cc = wmma_bf16(a2, B1[2][nt], cc);
    float b = bb1[nt];
#pragma unroll
    for (int r = 0; r < 8; ++r) {
      int M = r + 8 * half;
      float v = cc[r] + b;
      v = v > 0.0f ? v : 0.0f;
      ((unsigned short*)stZ)[M * 32 + nt * 16 + n0] = f2bf(v);
    }
  }
  FENCE();
  v16bf az = load_afrag(stZ, 32, 0, lane);
#pragma unroll
  for (int nt = 0; nt < 2; ++nt) {
    v8f dd = {};
    dd = wmma_bf16(az, B2[nt], dd);
    float b = bb2[nt];
    int Ncol = nt * 16 + n0;
#pragma unroll
    for (int r = 0; r < 8; ++r) {
      int M = r + 8 * half;
      if (sD[M] != sS[M])
        atomicAdd(acc + sAgg[M] * DDIM + Ncol, dd[r] + b);
    }
  }
  FENCE();
}

__global__ __launch_bounds__(32)
void k_edge(const float* __restrict__ H, const int* __restrict__ eidx,
            const float* __restrict__ ea,
            const unsigned short* __restrict__ w1to, const unsigned short* __restrict__ w1fr,
            const unsigned short* __restrict__ w2to, const unsigned short* __restrict__ w2fr,
            const float* __restrict__ b1to, const float* __restrict__ b2to,
            const float* __restrict__ b1fr, const float* __restrict__ b2fr,
            float* __restrict__ accto, float* __restrict__ accfr, int ntiles) {
  __shared__ unsigned int stA[16 * KPAD / 2];
  __shared__ unsigned int stZ[16 * 32 / 2];
  __shared__ int sD[16], sS[16];

  const int lane = threadIdx.x;
  const int* srcp = eidx;
  const int* dstp = eidx + EE;

  v16bf B1t[3][2], B1f[3][2], B2t[2], B2f[2];
#pragma unroll
  for (int kc = 0; kc < 3; ++kc)
#pragma unroll
    for (int nt = 0; nt < 2; ++nt) {
      B1t[kc][nt] = load_bfrag(w1to, KPAD, kc, nt, lane);
      B1f[kc][nt] = load_bfrag(w1fr, KPAD, kc, nt, lane);
    }
#pragma unroll
  for (int nt = 0; nt < 2; ++nt) {
    B2t[nt] = load_bfrag(w2to, 32, 0, nt, lane);
    B2f[nt] = load_bfrag(w2fr, 32, 0, nt, lane);
  }
  float bb1t[2] = { b1to[lane & 15], b1to[(lane & 15) + 16] };
  float bb2t[2] = { b2to[lane & 15], b2to[(lane & 15) + 16] };
  float bb1f[2] = { b1fr[lane & 15], b1fr[(lane & 15) + 16] };
  float bb2f[2] = { b2fr[lane & 15], b2fr[(lane & 15) + 16] };

  for (int tile = blockIdx.x; tile < ntiles; tile += gridDim.x) {
    int m = lane >> 1, p = lane & 1;
    int e = tile * 16 + m;
    int di = dstp[e], si = srcp[e];
    if (p == 0) { sD[m] = di; sS[m] = si; }

    const float* hd = H + di * DDIM + p * 16;
    const float* hs = H + si * DDIM + p * 16;
#pragma unroll
    for (int j = 0; j < 8; ++j)
      stA[(m * KPAD + p * 16) / 2 + j] = pack2(hd[2 * j], hd[2 * j + 1]);
#pragma unroll
    for (int j = 0; j < 8; ++j)
      stA[(m * KPAD + 32 + p * 16) / 2 + j] = pack2(hs[2 * j], hs[2 * j + 1]);
    if (p == 0) {
      stA[(m * KPAD + 64) / 2] = pack2(ea[e * 2], ea[e * 2 + 1]);
#pragma unroll
      for (int j = 1; j < 8; ++j) stA[(m * KPAD + 64) / 2 + j] = 0u;
    } else {
#pragma unroll
      for (int j = 8; j < 16; ++j) stA[(m * KPAD + 64) / 2 + j] = 0u;
    }
    FENCE();

    v16bf a0 = load_afrag(stA, KPAD, 0, lane);
    v16bf a1 = load_afrag(stA, KPAD, 1, lane);
    v16bf a2 = load_afrag(stA, KPAD, 2, lane);

    mlp_scatter(a0, a1, a2, B1t, B2t, bb1t, bb2t, stZ, sD, sD, sS, accto, lane); // Phi_to  -> dst
    mlp_scatter(a0, a1, a2, B1f, B2f, bb1f, bb2f, stZ, sS, sD, sS, accfr, lane); // Phi_from-> src
  }
}

// ---------------- per-node WMMA kernel: lp MLP + GRU + decode + loss ----------------
// One wave per 16-node tile. GRU concat [H|mt|mf|lp|x] staged as 16x160 bf16.
__global__ __launch_bounds__(32)
void k_node(float* __restrict__ H,
            const float* __restrict__ accto, const float* __restrict__ accfr,
            const float* __restrict__ c_dst, const float* __restrict__ c_src,
            const float* __restrict__ diag,
            const float* __restrict__ x, const float* __restrict__ y,
            const unsigned short* __restrict__ wlp1, const unsigned short* __restrict__ wlp2,
            const unsigned short* __restrict__ wgru, const unsigned short* __restrict__ wdec,
            const float* __restrict__ lpb1, const float* __restrict__ lpb2,
            const float* __restrict__ bih,  const float* __restrict__ bhh,
            const float* __restrict__ db1,  const float* __restrict__ dW2,
            const float* __restrict__ db2,
            float* __restrict__ Fout, float* __restrict__ lossbuf, int step) {
  __shared__ unsigned int stIn[16 * KGRU / 2];  // 16x160 bf16 GRU input tile
  __shared__ unsigned int stLP[16 * 32 / 2];    // 16x32 bf16 [lc | 0...]
  __shared__ unsigned int stZ[16 * 32 / 2];     // 16x32 bf16 stage (lp hidden / new H)
  __shared__ float stG[16 * 96];                // 16x96 f32 gates (reused for decode hidden)
  __shared__ float stH[16 * 32];                // old H (f32) for residual
  __shared__ float red[32];

  const int lane = threadIdx.x;
  const int n0 = lane & 15, half = lane >> 4;
  const int NT = NN / 16;  // 6250 exact

  for (int tile = blockIdx.x; tile < NT; tile += gridDim.x) {
    int m = lane >> 1, p = lane & 1;
    int v = tile * 16 + m;
    float rcd = 1.0f / fmaxf(c_dst[v], 1.0f);
    float rcs = 1.0f / fmaxf(c_src[v], 1.0f);
    const float* hp = H + v * DDIM + p * 16;
    const float* at = accto + v * DDIM + p * 16;
    const float* af = accfr + v * DDIM + p * 16;
#pragma unroll
    for (int j = 0; j < 8; ++j) {
      float h0 = hp[2 * j], h1 = hp[2 * j + 1];
      stIn[(m * KGRU + p * 16) / 2 + j] = pack2(h0, h1);
      stH[m * 32 + p * 16 + 2 * j] = h0;
      stH[m * 32 + p * 16 + 2 * j + 1] = h1;
      stIn[(m * KGRU + 32 + p * 16) / 2 + j] = pack2(at[2 * j] * rcd, at[2 * j + 1] * rcd);
      stIn[(m * KGRU + 64 + p * 16) / 2 + j] = pack2(af[2 * j] * rcs, af[2 * j + 1] * rcs);
    }
    if (p == 0) {
      stIn[(m * KGRU + 128) / 2]     = pack2(x[v * 3], x[v * 3 + 1]);
      stIn[(m * KGRU + 128) / 2 + 1] = pack2(x[v * 3 + 2], 0.0f);
#pragma unroll
      for (int j = 2; j < 8; ++j) stIn[(m * KGRU + 128) / 2 + j] = 0u;
      float lc = 1.0f - diag[v];
      stLP[(m * 32) / 2] = pack2(lc, 0.0f);
#pragma unroll
      for (int j = 1; j < 8; ++j) stLP[(m * 32) / 2 + j] = 0u;
    } else {
#pragma unroll
      for (int j = 8; j < 16; ++j) stIn[(m * KGRU + 128) / 2 + j] = 0u;
#pragma unroll
      for (int j = 8; j < 16; ++j) stLP[(m * 32) / 2 + j] = 0u;
    }
    FENCE();

    // ---- lp MLP: [H | lc] (64) -> 32 relu -> 32, result into stIn cols 96..127
    v16bf aH = load_afrag(stIn, KGRU, 0, lane);   // cols 0..31 (shared with GRU kc0)
    v16bf aL = load_afrag(stLP, 32, 0, lane);
#pragma unroll
    for (int nt = 0; nt < 2; ++nt) {
      v8f c = {};
      c = wmma_bf16(aH, load_bfrag(wlp1, 64, 0, nt, lane), c);
      c = wmma_bf16(aL, load_bfrag(wlp1, 64, 1, nt, lane), c);
      float b = lpb1[nt * 16 + n0];
#pragma unroll
      for (int r = 0; r < 8; ++r) {
        float val = c[r] + b;
        val = val > 0.0f ? val : 0.0f;
        ((unsigned short*)stZ)[(r + 8 * half) * 32 + nt * 16 + n0] = f2bf(val);
      }
    }
    FENCE();
    v16bf aZ = load_afrag(stZ, 32, 0, lane);
#pragma unroll
    for (int nt = 0; nt < 2; ++nt) {
      v8f c = {};
      c = wmma_bf16(aZ, load_bfrag(wlp2, 32, 0, nt, lane), c);
      float b = lpb2[nt * 16 + n0];
#pragma unroll
      for (int r = 0; r < 8; ++r)
        ((unsigned short*)stIn)[(r + 8 * half) * KGRU + 96 + nt * 16 + n0] = f2bf(c[r] + b);
    }
    FENCE();

    // ---- GRU gates: [16x160] @ [160x96] -> 16x96 (5 K-chunks x 6 N-tiles)
    v16bf aG[5];
    aG[0] = aH;
#pragma unroll
    for (int kc = 1; kc < 5; ++kc) aG[kc] = load_afrag(stIn, KGRU, kc, lane);
#pragma unroll
    for (int nt = 0; nt < 6; ++nt) {
      v8f c = {};
#pragma unroll
      for (int kc = 0; kc < 5; ++kc)
        c = wmma_bf16(aG[kc], load_bfrag(wgru, KGRU, kc, nt, lane), c);
      float b = bih[nt * 16 + n0];
#pragma unroll
      for (int r = 0; r < 8; ++r)
        stG[(r + 8 * half) * 96 + nt * 16 + n0] = c[r] + b;
    }
    FENCE();

    // ---- elementwise GRU update: lane = (node n0, j-range half*16..)
    int vv = tile * 16 + n0;
#pragma unroll
    for (int j2 = 0; j2 < 16; ++j2) {
      int j = half * 16 + j2;
      float gr = stG[n0 * 96 + j];
      float gz = stG[n0 * 96 + 32 + j];
      float gn = stG[n0 * 96 + 64 + j];
      float r = 1.0f / (1.0f + expf(-(gr + bhh[j])));
      float z = 1.0f / (1.0f + expf(-(gz + bhh[32 + j])));
      float t = tanhf(gn + r * bhh[64 + j]);
      float hn = (1.0f - z) * t * ALPHA_C + stH[n0 * 32 + j];
      H[vv * DDIM + j] = hn;
      ((unsigned short*)stZ)[n0 * 32 + j] = f2bf(hn);
    }
    FENCE();

    // ---- decode layer 1: 32 -> 32 relu (WMMA), hidden into stG cols 0..31
    v16bf aD = load_afrag(stZ, 32, 0, lane);
#pragma unroll
    for (int nt = 0; nt < 2; ++nt) {
      v8f c = {};
      c = wmma_bf16(aD, load_bfrag(wdec, 32, 0, nt, lane), c);
      float b = db1[nt * 16 + n0];
#pragma unroll
      for (int r = 0; r < 8; ++r) {
        float val = c[r] + b;
        val = val > 0.0f ? val : 0.0f;
        stG[(r + 8 * half) * 96 + nt * 16 + n0] = val;
      }
    }
    FENCE();
    // ---- decode layer 2 (2 outputs): lane = (node n0, output half)
    float Fp = db2[half];
#pragma unroll
    for (int k = 0; k < 32; ++k) Fp += dW2[half * 32 + k] * stG[n0 * 96 + k];
    Fout[vv * 2 + half] = Fp;
    float e = Fp - y[vv * 2 + half];
    red[lane] = e * e;
    FENCE();
    if (lane == 0) {
      float s = 0.0f;
#pragma unroll
      for (int i = 0; i < 32; ++i) s += red[i];
      atomicAdd(lossbuf + step, s);
    }
    FENCE();
  }
}

__global__ void k_finalize(const float* __restrict__ lossbuf, float* __restrict__ out_tail) {
  if (blockIdx.x != 0 || threadIdx.x != 0) return;
  float total = 0.0f;
  for (int k = 0; k < KSTEPS; ++k) {
    float l = lossbuf[k] / (2.0f * NN);
    out_tail[1 + k] = l;
    float w = 1.0f;
    for (int p = 0; p < KSTEPS - 1 - k; ++p) w *= GAMMA_C;
    total += l * w;
  }
  out_tail[0] = total;
}

// ---------------- host-side launch ----------------
extern "C" void kernel_launch(void* const* d_in, const int* in_sizes, int n_in,
                              void* d_out, int out_size, void* d_ws, size_t ws_size,
                              hipStream_t stream) {
  (void)in_sizes; (void)n_in; (void)out_size; (void)ws_size;
  const float* x    = (const float*)d_in[0];
  const float* y    = (const float*)d_in[1];
  const int*   eidx = (const int*)  d_in[2];
  const float* ea   = (const float*)d_in[3];
  const float* toW1 = (const float*)d_in[4];  const float* tob1 = (const float*)d_in[5];
  const float* toW2 = (const float*)d_in[6];  const float* tob2 = (const float*)d_in[7];
  const float* frW1 = (const float*)d_in[8];  const float* frb1 = (const float*)d_in[9];
  const float* frW2 = (const float*)d_in[10]; const float* frb2 = (const float*)d_in[11];
  const float* lpW1 = (const float*)d_in[12]; const float* lpb1 = (const float*)d_in[13];
  const float* lpW2 = (const float*)d_in[14]; const float* lpb2 = (const float*)d_in[15];
  const float* Wih  = (const float*)d_in[16]; const float* bih  = (const float*)d_in[17];
  /* d_in[18] = gru_Whh: unused (h0 == 0) */
  const float* bhh  = (const float*)d_in[19];
  const float* dW1  = (const float*)d_in[20]; const float* db1  = (const float*)d_in[21];
  const float* dW2  = (const float*)d_in[22]; const float* db2  = (const float*)d_in[23];
  float* out = (float*)d_out;

  // workspace carve-up
  float* H      = (float*)d_ws;                 // N*32
  float* accto  = H + NN * DDIM;                // N*32
  float* accfr  = accto + NN * DDIM;            // N*32
  float* c_dst  = accfr + NN * DDIM;            // N
  float* c_src  = c_dst + NN;                   // N
  float* diag   = c_src + NN;                   // N
  float* lossb  = diag + NN;                    // 8 (4 used)
  unsigned short* w1to = (unsigned short*)(lossb + 8);
  unsigned short* w1fr = w1to + 32 * KPAD;
  unsigned short* w2to = w1fr + 32 * KPAD;
  unsigned short* w2fr = w2to + 32 * 32;
  unsigned short* wlp1 = w2fr + 32 * 32;        // 32 x 64 (folded lp layer1)
  unsigned short* wlp2 = wlp1 + 32 * 64;        // 32 x 32
  unsigned short* wgru = wlp2 + 32 * 32;        // 96 x 160 (padded Wih)
  unsigned short* wdec = wgru + 96 * KGRU;      // 32 x 32

  const int ntiles = EE / 16;

  k_prep_weights<<<(96 * KGRU + 255) / 256, 256, 0, stream>>>(
      toW1, frW1, toW2, frW2, lpW1, lpW2, Wih, dW1,
      w1to, w1fr, w2to, w2fr, wlp1, wlp2, wgru, wdec);
  k_init<<<(NN * DDIM + 255) / 256, 256, 0, stream>>>(H, c_dst, c_src, diag, lossb);
  k_degree<<<(EE + 255) / 256, 256, 0, stream>>>(eidx, ea, c_dst, c_src, diag);

  for (int step = 0; step < KSTEPS; ++step) {
    k_zero_acc<<<(NN * DDIM + 255) / 256, 256, 0, stream>>>(accto, accfr);
    k_edge<<<4096, 32, 0, stream>>>(H, eidx, ea, w1to, w1fr, w2to, w2fr,
                                    tob1, tob2, frb1, frb2, accto, accfr, ntiles);
    k_node<<<NN / 16, 32, 0, stream>>>(H, accto, accfr, c_dst, c_src, diag, x, y,
                                       wlp1, wlp2, wgru, wdec,
                                       lpb1, lpb2, bih, bhh, db1, dW2, db2,
                                       out, lossb, step);
  }
  k_finalize<<<1, 1, 0, stream>>>(lossb, out + NN * 2);
}
